// SimpleMoELayer_25314537242700
// MI455X (gfx1250) — matrix operand order repaired and checked by reading, not compiled, not run on hardware
//
#include <hip/hip_runtime.h>
#include <hip/hip_bf16.h>

// ---------------------------------------------------------------------------
// MoE transformer block for MI455X (gfx1250, wave32, WMMA bf16 16x16x32).
// Sparse top-2 expert routing (4x fewer FLOPs than the dense reference),
// flash attention, fp32->bf16 LDS transpose staging for all weight tiles.
// Requires ws_size >= ~128 MB.
// ---------------------------------------------------------------------------

#define H    1024
#define S    2048
#define Bd   2
#define T    (Bd * S)       // 4096 tokens
#define NH   16
#define HD   64
#define E    8
#define Isz  2048
#define RMAX 9216           // 8192 selected rows + per-expert pad to 128
#define NBLK 72             // RMAX / 128

typedef __bf16 bf16_t;
typedef __attribute__((ext_vector_type(16))) __bf16        v16bf;
typedef __attribute__((ext_vector_type(8)))  float         v8f;
typedef __attribute__((ext_vector_type(4)))  unsigned int  u32x4;
typedef __attribute__((ext_vector_type(4)))  float         f32x4;

union FragU { v16bf v; u32x4 q[2]; };

__device__ __forceinline__ v8f wmma_bf16(v16bf a, v16bf b, v8f c) {
  return __builtin_amdgcn_wmma_f32_16x16x32_bf16(false, a, false, b, (short)0, c,
                                                 false, false);
}

__device__ __forceinline__ void zero8(v8f& x) {
#pragma unroll
  for (int i = 0; i < 8; ++i) x[i] = 0.f;
}

// A-fragment (16x32 bf16, row-major source, lane = row l%16):
// contiguous 16 k-halves at (l/16)*16
__device__ __forceinline__ v16bf load_afrag(const bf16_t* rowk, int lane) {
  FragU f;
  const bf16_t* p = rowk + ((lane >> 4) << 4);
  f.q[0] = *(const u32x4*)p;
  f.q[1] = *(const u32x4*)(p + 8);
  return f.v;
}

// B-fragment from LDS tile stored transposed Bt[n][k], pitch 40 halves.
// lane holds column n = ncol + l%16, contiguous 16 k-halves at (l/16)*16.
__device__ __forceinline__ v16bf load_bfrag_lds(const bf16_t* bt, int ncol, int lane) {
  FragU f;
  const bf16_t* p = bt + (ncol + (lane & 15)) * 40 + ((lane >> 4) << 4);
  f.q[0] = *(const u32x4*)p;
  f.q[1] = *(const u32x4*)(p + 8);
  return f.v;
}

// Stage a K32 x N64 fp32 weight tile -> bf16 transposed LDS tile (pitch 40).
// 256 threads: thread t handles row k=t/8, 8 cols starting at (t%8)*8.
__device__ __forceinline__ void stage_b(bf16_t* bt, const float* src0, int ldb, int tid) {
  int k  = tid >> 3;
  int nb = (tid & 7) << 3;
  const float* src = src0 + (size_t)k * ldb + nb;
  f32x4 a = *(const f32x4*)src;
  f32x4 b = *(const f32x4*)(src + 4);
  __builtin_prefetch(src + 32 * (size_t)ldb, 0, 1);   // next k-tile -> global_prefetch
#pragma unroll
  for (int i = 0; i < 4; ++i) bt[(nb + i) * 40 + k] = (bf16_t)a[i];
#pragma unroll
  for (int i = 0; i < 4; ++i) bt[(nb + 4 + i) * 40 + k] = (bf16_t)b[i];
}

// ---------------------------------------------------------------------------
// RMSNorm: one block per token, 256 threads x 4 floats, bf16 output
// ---------------------------------------------------------------------------
__global__ __launch_bounds__(256) void k_rmsnorm(const float* __restrict__ x,
                                                 const float* __restrict__ w,
                                                 bf16_t* __restrict__ out) {
  __shared__ float red[8];
  int t = blockIdx.x, tid = threadIdx.x;
  const float* xr = x + (size_t)t * H + tid * 4;
  f32x4 v = *(const f32x4*)xr;
  float ss = v[0]*v[0] + v[1]*v[1] + v[2]*v[2] + v[3]*v[3];
#pragma unroll
  for (int m = 16; m >= 1; m >>= 1) ss += __shfl_xor(ss, m, 32);
  if ((tid & 31) == 0) red[tid >> 5] = ss;
  __syncthreads();
  float tot = 0.f;
#pragma unroll
  for (int i = 0; i < 8; ++i) tot += red[i];
  float inv = rsqrtf(tot * (1.f / H) + 1e-6f);
  f32x4 wv = *(const f32x4*)(w + tid * 4);
  bf16_t* o = out + (size_t)t * H + tid * 4;
#pragma unroll
  for (int i = 0; i < 4; ++i) o[i] = (bf16_t)(v[i] * inv * wv[i]);
}

// ---------------------------------------------------------------------------
// Fused QKV GEMM: C[T,H] = A(bf16) x W(fp32), block tile 128x64, 8 waves
// ---------------------------------------------------------------------------
__global__ __launch_bounds__(256) void k_qkv(const bf16_t* __restrict__ xn,
                                             const float* __restrict__ wq,
                                             const float* __restrict__ wk,
                                             const float* __restrict__ wv,
                                             bf16_t* __restrict__ qo,
                                             bf16_t* __restrict__ ko,
                                             bf16_t* __restrict__ vo) {
  __shared__ alignas(16) bf16_t bt[64 * 40];
  const float* W = blockIdx.z == 0 ? wq : blockIdx.z == 1 ? wk : wv;
  bf16_t*      O = blockIdx.z == 0 ? qo : blockIdx.z == 1 ? ko : vo;
  int tid = threadIdx.x, lane = tid & 31, w = tid >> 5;
  int row0 = blockIdx.x * 128 + w * 16;
  int col0 = blockIdx.y * 64;
  int m16 = lane & 15, hi = lane >> 4;
  const bf16_t* arow = xn + (size_t)(row0 + m16) * H;
  v8f acc[4];
#pragma unroll
  for (int j = 0; j < 4; ++j) zero8(acc[j]);
  for (int kt = 0; kt < H; kt += 32) {
    stage_b(bt, W + (size_t)kt * H + col0, H, tid);
    __syncthreads();
    v16bf af = load_afrag(arow + kt, lane);
#pragma unroll
    for (int j = 0; j < 4; ++j)
      acc[j] = wmma_bf16(af, load_bfrag_lds(bt, j * 16, lane), acc[j]);
    __syncthreads();
  }
#pragma unroll
  for (int j = 0; j < 4; ++j)
#pragma unroll
    for (int r = 0; r < 8; ++r)
      O[(size_t)(row0 + hi * 8 + r) * H + col0 + j * 16 + m16] = (bf16_t)acc[j][r];
}

// ---------------------------------------------------------------------------
// WO GEMM + residual add, fp32 output straight into d_out
// ---------------------------------------------------------------------------
__global__ __launch_bounds__(256) void k_wo(const bf16_t* __restrict__ a,
                                            const float* __restrict__ wo,
                                            const float* __restrict__ resid,
                                            float* __restrict__ out) {
  __shared__ alignas(16) bf16_t bt[64 * 40];
  int tid = threadIdx.x, lane = tid & 31, w = tid >> 5;
  int row0 = blockIdx.x * 128 + w * 16;
  int col0 = blockIdx.y * 64;
  int m16 = lane & 15, hi = lane >> 4;
  const bf16_t* arow = a + (size_t)(row0 + m16) * H;
  v8f acc[4];
#pragma unroll
  for (int j = 0; j < 4; ++j) zero8(acc[j]);
  for (int kt = 0; kt < H; kt += 32) {
    stage_b(bt, wo + (size_t)kt * H + col0, H, tid);
    __syncthreads();
    v16bf af = load_afrag(arow + kt, lane);
#pragma unroll
    for (int j = 0; j < 4; ++j)
      acc[j] = wmma_bf16(af, load_bfrag_lds(bt, j * 16, lane), acc[j]);
    __syncthreads();
  }
#pragma unroll
  for (int j = 0; j < 4; ++j)
#pragma unroll
    for (int r = 0; r < 8; ++r) {
      size_t idx = (size_t)(row0 + hi * 8 + r) * H + col0 + j * 16 + m16;
      out[idx] = resid[idx] + acc[j][r];
    }
}

// ---------------------------------------------------------------------------
// Flash attention: block = 4 waves = 64 queries of one (batch, head);
// per 32-key step: 4 score WMMAs, online softmax, 4 PV WMMAs.
// ---------------------------------------------------------------------------
__global__ __launch_bounds__(128) void k_attn(const bf16_t* __restrict__ Q,
                                              const bf16_t* __restrict__ K,
                                              const bf16_t* __restrict__ V,
                                              bf16_t* __restrict__ O) {
  __shared__ alignas(16) bf16_t vt[64 * 40];       // V^T tile: [d][key]
  __shared__ alignas(16) bf16_t pa[4][16 * 40];    // per-wave P tile (A layout)
  int tid = threadIdx.x, lane = tid & 31, w = tid >> 5;
  int m16 = lane & 15, hi = lane >> 4;
  int q0 = blockIdx.x * 64 + w * 16;
  size_t base = (size_t)blockIdx.z * S * H + (size_t)blockIdx.y * HD;
  const bf16_t* qrow = Q + base + (size_t)(q0 + m16) * H;
  v16bf qf0 = load_afrag(qrow, lane);
  v16bf qf1 = load_afrag(qrow + 32, lane);
  v8f o[4];
#pragma unroll
  for (int j = 0; j < 4; ++j) zero8(o[j]);
  float mrow[8], lrow[8];
#pragma unroll
  for (int r = 0; r < 8; ++r) { mrow[r] = -1e30f; lrow[r] = 0.f; }

  for (int kk0 = 0; kk0 < S; kk0 += 32) {
    __syncthreads();
    { // cooperatively transpose 32 keys x 64 dims of V into LDS
      int kkl = tid & 31, d0 = (tid >> 5) << 4;
      const bf16_t* vr = V + base + (size_t)(kk0 + kkl) * H + d0;
      FragU tmp;
      tmp.q[0] = *(const u32x4*)vr;
      tmp.q[1] = *(const u32x4*)(vr + 8);
#pragma unroll
      for (int i = 0; i < 16; ++i) vt[(d0 + i) * 40 + kkl] = tmp.v[i];
    }
    __syncthreads();

    v8f s0, s1; zero8(s0); zero8(s1);
    {
      const bf16_t* kr = K + base + (size_t)(kk0 + m16) * H + (hi << 4);
      FragU k0a, k0b, k1a, k1b;
      k0a.q[0] = *(const u32x4*)kr;        k0a.q[1] = *(const u32x4*)(kr + 8);
      k0b.q[0] = *(const u32x4*)(kr + 32); k0b.q[1] = *(const u32x4*)(kr + 40);
      s0 = wmma_bf16(qf0, k0a.v, s0);
      s0 = wmma_bf16(qf1, k0b.v, s0);
      const bf16_t* kr1 = kr + 16 * H;
      k1a.q[0] = *(const u32x4*)kr1;        k1a.q[1] = *(const u32x4*)(kr1 + 8);
      k1b.q[0] = *(const u32x4*)(kr1 + 32); k1b.q[1] = *(const u32x4*)(kr1 + 40);
      s1 = wmma_bf16(qf0, k1a.v, s1);
      s1 = wmma_bf16(qf1, k1b.v, s1);
    }
    s0 = s0 * 0.125f;   // 1/sqrt(HD)
    s1 = s1 * 0.125f;

    bf16_t* paw = pa[w];
#pragma unroll
    for (int r = 0; r < 8; ++r) {
      float mx = fmaxf(s0[r], s1[r]);
#pragma unroll
      for (int msk = 1; msk < 16; msk <<= 1) mx = fmaxf(mx, __shfl_xor(mx, msk, 32));
      float mnew  = fmaxf(mrow[r], mx);
      float alpha = __expf(mrow[r] - mnew);
      float p0 = __expf(s0[r] - mnew);
      float p1 = __expf(s1[r] - mnew);
      float rs = p0 + p1;
#pragma unroll
      for (int msk = 1; msk < 16; msk <<= 1) rs += __shfl_xor(rs, msk, 32);
      lrow[r] = lrow[r] * alpha + rs;
      mrow[r] = mnew;
#pragma unroll
      for (int j = 0; j < 4; ++j) o[j][r] *= alpha;
      paw[(r + hi * 8) * 40 + m16]      = (bf16_t)p0;
      paw[(r + hi * 8) * 40 + 16 + m16] = (bf16_t)p1;
    }
    // reload P in A-fragment layout (same-wave DS ops are in-order)
    FragU pf;
    const bf16_t* pp = paw + m16 * 40 + hi * 16;
    pf.q[0] = *(const u32x4*)pp;
    pf.q[1] = *(const u32x4*)(pp + 8);
#pragma unroll
    for (int jn = 0; jn < 4; ++jn) {
      FragU vf;
      const bf16_t* vp = vt + (jn * 16 + m16) * 40 + (hi << 4);
      vf.q[0] = *(const u32x4*)vp;
      vf.q[1] = *(const u32x4*)(vp + 8);
      o[jn] = wmma_bf16(pf.v, vf.v, o[jn]);
    }
  }
#pragma unroll
  for (int jn = 0; jn < 4; ++jn)
#pragma unroll
    for (int r = 0; r < 8; ++r) {
      float val = o[jn][r] / lrow[r];
      O[base + (size_t)(q0 + hi * 8 + r) * H + jn * 16 + m16] = (bf16_t)val;
    }
}

// ---------------------------------------------------------------------------
// Router: one wave per token; recompute rmsnorm in fp32, 8 logits, top-2
// ---------------------------------------------------------------------------
__global__ __launch_bounds__(32) void k_router(const float* __restrict__ hidden,
                                               const float* __restrict__ wpost,
                                               const float* __restrict__ wgate,
                                               float* __restrict__ logits,
                                               float* __restrict__ rwv,
                                               int* __restrict__ selv,
                                               int* __restrict__ counts) {
  int t = blockIdx.x, lane = threadIdx.x;
  const float* xr = hidden + (size_t)t * H;
  float ss = 0.f;
  for (int h = lane; h < H; h += 32) { float x = xr[h]; ss += x * x; }
#pragma unroll
  for (int m = 16; m >= 1; m >>= 1) ss += __shfl_xor(ss, m, 32);
  float inv = rsqrtf(ss * (1.f / H) + 1e-6f);
  float acc[E];
#pragma unroll
  for (int e = 0; e < E; ++e) acc[e] = 0.f;
  for (int h = lane; h < H; h += 32) {
    float x = xr[h] * inv * wpost[h];
    const float* g = wgate + (size_t)h * E;
    f32x4 g0 = *(const f32x4*)g;
    f32x4 g1 = *(const f32x4*)(g + 4);
    acc[0] += x * g0[0]; acc[1] += x * g0[1]; acc[2] += x * g0[2]; acc[3] += x * g0[3];
    acc[4] += x * g1[0]; acc[5] += x * g1[1]; acc[6] += x * g1[2]; acc[7] += x * g1[3];
  }
#pragma unroll
  for (int e = 0; e < E; ++e)
#pragma unroll
    for (int m = 16; m >= 1; m >>= 1) acc[e] += __shfl_xor(acc[e], m, 32);
  if (lane == 0) {
    float* lo = logits + (size_t)t * E;
    float mx = acc[0];
#pragma unroll
    for (int e = 1; e < E; ++e) mx = fmaxf(mx, acc[e]);
    float p[E]; float sum = 0.f;
#pragma unroll
    for (int e = 0; e < E; ++e) { lo[e] = acc[e]; p[e] = __expf(acc[e] - mx); sum += p[e]; }
    float rinv = 1.f / sum;
#pragma unroll
    for (int e = 0; e < E; ++e) p[e] *= rinv;
    int e1 = 0;
#pragma unroll
    for (int e = 1; e < E; ++e) if (p[e] > p[e1]) e1 = e;   // ties -> lower index
    int e2 = -1;
#pragma unroll
    for (int e = 0; e < E; ++e)
      if (e != e1 && (e2 < 0 || p[e] > p[e2])) e2 = e;
    rwv[t * 2]     = p[e1];  rwv[t * 2 + 1]  = p[e2];
    selv[t * 2]    = e1;     selv[t * 2 + 1] = e2;
    atomicAdd(&counts[e1], 1);
    atomicAdd(&counts[e2], 1);
  }
}

// ---------------------------------------------------------------------------
// Routing bookkeeping
// ---------------------------------------------------------------------------
__global__ __launch_bounds__(256) void k_init(int* perm, int* counts, int* cursors) {
  int i = blockIdx.x * 256 + threadIdx.x;
  if (i < RMAX) perm[i] = 0;
  if (i < E) { counts[i] = 0; cursors[i] = 0; }
}

__global__ void k_scan(const int* counts, int* offs, int* blk_e) {
  if (threadIdx.x == 0 && blockIdx.x == 0) {
    int off = 0, bi = 0;
    for (int e = 0; e < E; ++e) {
      offs[e] = off;
      int nb = (counts[e] + 127) >> 7;
      for (int i = 0; i < nb; ++i) blk_e[bi++] = e;
      off += nb << 7;
    }
    for (; bi < NBLK; ++bi) blk_e[bi] = -1;
  }
}

__global__ __launch_bounds__(256) void k_scatter(const int* __restrict__ selv,
                                                 const int* __restrict__ offs,
                                                 int* __restrict__ cursors,
                                                 int* __restrict__ perm,
                                                 int* __restrict__ rowidx) {
  int t = blockIdx.x * 256 + threadIdx.x;
  if (t >= T) return;
#pragma unroll
  for (int j = 0; j < 2; ++j) {
    int e = selv[t * 2 + j];
    int pos = atomicAdd(&cursors[e], 1);
    int row = offs[e] + pos;
    perm[row] = t;
    rowidx[t * 2 + j] = row;
  }
}

// ---------------------------------------------------------------------------
// Grouped expert gate+up GEMM fused with silu(g)*u, bf16 out (128-row blocks)
// ---------------------------------------------------------------------------
__global__ __launch_bounds__(256) void k_gateup(const bf16_t* __restrict__ xm,
                                                const float* __restrict__ wg,
                                                const float* __restrict__ wu,
                                                const int* __restrict__ perm,
                                                const int* __restrict__ blk_e,
                                                bf16_t* __restrict__ act) {
  __shared__ alignas(16) bf16_t lg[64 * 40];
  __shared__ alignas(16) bf16_t lu[64 * 40];
  int e = blk_e[blockIdx.x];
  if (e < 0) return;
  int tid = threadIdx.x, lane = tid & 31, w = tid >> 5;
  int row0 = blockIdx.x * 128 + w * 16;
  int col0 = blockIdx.y * 64;
  int m16 = lane & 15, hi = lane >> 4;
  int tok = perm[row0 + m16];
  const bf16_t* arow = xm + (size_t)tok * H;
  const float* Wg = wg + (size_t)e * H * Isz + col0;
  const float* Wu = wu + (size_t)e * H * Isz + col0;
  v8f ag[4], au[4];
#pragma unroll
  for (int j = 0; j < 4; ++j) { zero8(ag[j]); zero8(au[j]); }
  for (int kt = 0; kt < H; kt += 32) {
    stage_b(lg, Wg + (size_t)kt * Isz, Isz, tid);
    stage_b(lu, Wu + (size_t)kt * Isz, Isz, tid);
    __syncthreads();
    v16bf af = load_afrag(arow + kt, lane);
#pragma unroll
    for (int j = 0; j < 4; ++j) {
      ag[j] = wmma_bf16(af, load_bfrag_lds(lg, j * 16, lane), ag[j]);
      au[j] = wmma_bf16(af, load_bfrag_lds(lu, j * 16, lane), au[j]);
    }
    __syncthreads();
  }
#pragma unroll
  for (int j = 0; j < 4; ++j)
#pragma unroll
    for (int r = 0; r < 8; ++r) {
      float g = ag[j][r], u = au[j][r];
      float s = g / (1.f + __expf(-g));       // silu
      act[(size_t)(row0 + hi * 8 + r) * Isz + col0 + j * 16 + m16] = (bf16_t)(s * u);
    }
}

// ---------------------------------------------------------------------------
// Grouped expert down GEMM, fp32 out
// ---------------------------------------------------------------------------
__global__ __launch_bounds__(256) void k_down(const bf16_t* __restrict__ act,
                                              const float* __restrict__ wd,
                                              const int* __restrict__ blk_e,
                                              float* __restrict__ dout) {
  __shared__ alignas(16) bf16_t bt[64 * 40];
  int e = blk_e[blockIdx.x];
  if (e < 0) return;
  int tid = threadIdx.x, lane = tid & 31, w = tid >> 5;
  int row0 = blockIdx.x * 128 + w * 16;
  int col0 = blockIdx.y * 64;
  int m16 = lane & 15, hi = lane >> 4;
  const bf16_t* arow = act + (size_t)(row0 + m16) * Isz;
  const float* W = wd + (size_t)e * Isz * H + col0;
  v8f acc[4];
#pragma unroll
  for (int j = 0; j < 4; ++j) zero8(acc[j]);
  for (int kt = 0; kt < Isz; kt += 32) {
    stage_b(bt, W + (size_t)kt * H, H, tid);
    __syncthreads();
    v16bf af = load_afrag(arow + kt, lane);
#pragma unroll
    for (int j = 0; j < 4; ++j)
      acc[j] = wmma_bf16(af, load_bfrag_lds(bt, j * 16, lane), acc[j]);
    __syncthreads();
  }
#pragma unroll
  for (int j = 0; j < 4; ++j)
#pragma unroll
    for (int r = 0; r < 8; ++r)
      dout[(size_t)(row0 + hi * 8 + r) * H + col0 + j * 16 + m16] = acc[j][r];
}

// ---------------------------------------------------------------------------
// Combine: out[t] += w0*down[row0] + w1*down[row1]   (deterministic gather)
// ---------------------------------------------------------------------------
__global__ __launch_bounds__(256) void k_combine(float* __restrict__ out,
                                                 const float* __restrict__ down,
                                                 const int* __restrict__ rowidx,
                                                 const float* __restrict__ rwv) {
  int t = blockIdx.x;
  int c = threadIdx.x * 4;
  int r0 = rowidx[t * 2], r1 = rowidx[t * 2 + 1];
  float w0 = rwv[t * 2], w1 = rwv[t * 2 + 1];
  f32x4 o = *(const f32x4*)&out[(size_t)t * H + c];
  f32x4 a = *(const f32x4*)&down[(size_t)r0 * H + c];
  f32x4 b = *(const f32x4*)&down[(size_t)r1 * H + c];
  o = o + a * w0 + b * w1;
  *(f32x4*)&out[(size_t)t * H + c] = o;
}

// ---------------------------------------------------------------------------
extern "C" void kernel_launch(void* const* d_in, const int* in_sizes, int n_in,
                              void* d_out, int out_size, void* d_ws, size_t ws_size,
                              hipStream_t stream) {
  (void)in_sizes; (void)n_in; (void)out_size; (void)ws_size;
  const float* hs     = (const float*)d_in[0];
  const float* w_in   = (const float*)d_in[1];
  const float* w_post = (const float*)d_in[2];
  const float* wq     = (const float*)d_in[3];
  const float* wk     = (const float*)d_in[4];
  const float* wv     = (const float*)d_in[5];
  const float* wo     = (const float*)d_in[6];
  const float* wgate  = (const float*)d_in[7];
  const float* wgp    = (const float*)d_in[8];
  const float* wup    = (const float*)d_in[9];
  const float* wdp    = (const float*)d_in[10];

  float* out_hidden = (float*)d_out;
  float* out_logits = out_hidden + (size_t)T * H;

  char* p = (char*)d_ws;
  auto take = [&](size_t bytes) -> char* {
    char* r = p;
    p += (bytes + 255) & ~(size_t)255;
    return r;
  };
  bf16_t* xn    = (bf16_t*)take((size_t)T * H * 2);
  bf16_t* qb    = (bf16_t*)take((size_t)T * H * 2);
  bf16_t* kb    = (bf16_t*)take((size_t)T * H * 2);
  bf16_t* vb    = (bf16_t*)take((size_t)T * H * 2);
  bf16_t* ao    = (bf16_t*)take((size_t)T * H * 2);
  bf16_t* xm    = (bf16_t*)take((size_t)T * H * 2);
  bf16_t* act   = (bf16_t*)take((size_t)RMAX * Isz * 2);
  float*  dnout = (float*) take((size_t)RMAX * H * 4);
  float*  rwv   = (float*) take((size_t)T * 2 * 4);
  int* selv     = (int*)take((size_t)T * 2 * 4);
  int* rowidx   = (int*)take((size_t)T * 2 * 4);
  int* perm     = (int*)take((size_t)RMAX * 4);
  int* counts   = (int*)take(64 * 4);
  int* cursors  = (int*)take(64 * 4);
  int* offs     = (int*)take(64 * 4);
  int* blk_e    = (int*)take(NBLK * 4);

  k_init<<<dim3((RMAX + 255) / 256), dim3(256), 0, stream>>>(perm, counts, cursors);
  k_rmsnorm<<<dim3(T), dim3(256), 0, stream>>>(hs, w_in, xn);
  k_qkv<<<dim3(T / 128, H / 64, 3), dim3(256), 0, stream>>>(xn, wq, wk, wv, qb, kb, vb);
  k_attn<<<dim3(S / 64, NH, Bd), dim3(128), 0, stream>>>(qb, kb, vb, ao);
  k_wo<<<dim3(T / 128, H / 64), dim3(256), 0, stream>>>(ao, wo, hs, out_hidden);
  k_rmsnorm<<<dim3(T), dim3(256), 0, stream>>>(out_hidden, w_post, xm);
  k_router<<<dim3(T), dim3(32), 0, stream>>>(out_hidden, w_post, wgate, out_logits,
                                             rwv, selv, counts);
  k_scan<<<dim3(1), dim3(32), 0, stream>>>(counts, offs, blk_e);
  k_scatter<<<dim3(T / 256), dim3(256), 0, stream>>>(selv, offs, cursors, perm, rowidx);
  k_gateup<<<dim3(NBLK, Isz / 64), dim3(256), 0, stream>>>(xm, wgp, wup, perm, blk_e, act);
  k_down<<<dim3(NBLK, H / 64), dim3(256), 0, stream>>>(act, wdp, blk_e, dnout);
  k_combine<<<dim3(T), dim3(256), 0, stream>>>(out_hidden, dnout, rowidx, rwv);
}
